// EquivariantGNN_76991583748623
// MI455X (gfx1250) — compile-verified, matrix operand order checked by reading, not run on hardware
//
#include <hip/hip_runtime.h>
#include <hip/hip_bf16.h>

typedef __attribute__((ext_vector_type(16))) _Float16 v16h;
typedef __attribute__((ext_vector_type(8)))  _Float16 v8h;
typedef __attribute__((ext_vector_type(8)))  float    v8f;
typedef __attribute__((ext_vector_type(4)))  float    v4f;
typedef __attribute__((ext_vector_type(4)))  int      v4i;

#define D64 64
#define LAYERS 4

// ---- helpers ------------------------------------------------------------

__device__ __forceinline__ v8f wmma_f16(v16h a, v16h b, v8f c) {
    // D = A(16x32 f16) * B(32x16 f16) + C(16x16 f32)
    return __builtin_amdgcn_wmma_f32_16x16x32_f16(false, a, false, b, (short)0, c, false, false);
}

// Load one A/B fragment (16 halfs per lane) from f16 memory laid out row-major,
// base already points at row_start + k_offset (k_offset includes the +8 for hi lanes).
// Element i <-> k = (i&7) + (i>=8 ? 16 : 0).
__device__ __forceinline__ v16h load_frag_h16(const _Float16* base) {
    v8h lo = *(const v8h*)(base);
    v8h hi = *(const v8h*)(base + 16);
    v16h a;
#pragma unroll
    for (int i = 0; i < 8; ++i) { a[i] = lo[i]; a[i + 8] = hi[i]; }
    return a;
}

// Same fragment but source is f32 (converted on load).
__device__ __forceinline__ v16h load_frag_f32(const float* base) {
    v16h a;
#pragma unroll
    for (int i = 0; i < 8; ++i) {
        a[i]     = (_Float16)base[i];
        a[i + 8] = (_Float16)base[i + 16];
    }
    return a;
}

__device__ __forceinline__ v16h load_frag_f32_relu(const float* base) {
    v16h a;
#pragma unroll
    for (int i = 0; i < 8; ++i) {
        a[i]     = (_Float16)fmaxf(base[i], 0.f);
        a[i + 8] = (_Float16)fmaxf(base[i + 16], 0.f);
    }
    return a;
}

__device__ __forceinline__ void atomAddF(float* p, float v) {
    __hip_atomic_fetch_add(p, v, __ATOMIC_RELAXED, __HIP_MEMORY_SCOPE_AGENT);
}

// ---- per-edge squared distance -----------------------------------------

__global__ void k_d2(const float* __restrict__ pos, const int* __restrict__ src,
                     const int* __restrict__ dst, float* __restrict__ d2, int E) {
    int e = blockIdx.x * blockDim.x + threadIdx.x;
    if (e >= E) return;
    int s = src[e], d = dst[e];
    float dx = pos[s * 3 + 0] - pos[d * 3 + 0];
    float dy = pos[s * 3 + 1] - pos[d * 3 + 1];
    float dz = pos[s * 3 + 2] - pos[d * 3 + 2];
    d2[e] = dx * dx + dy * dy + dz * dz;
}

// ---- embed: h = [pos | emb[z]] @ Win ------------------------------------

__global__ void k_embed(const float* __restrict__ pos, const int* __restrict__ z,
                        const float* __restrict__ emb, const float* __restrict__ Win,
                        float* __restrict__ h, _Float16* __restrict__ h16, int N) {
    __shared__ alignas(16) _Float16 sWt[D64 * D64];
    int tid = threadIdx.x;
    for (int i = tid; i < D64 * D64; i += blockDim.x) {
        int n = i >> 6, k = i & 63;
        sWt[i] = (_Float16)Win[k * D64 + n];
    }
    __syncthreads();
    int wv = tid >> 5, lane = tid & 31;
    int tile = blockIdx.x * 4 + wv;
    if (tile * 16 >= N) return;
    int nbase = tile * 16;
    int l15 = lane & 15;
    int off = (lane >> 4) << 3;

    int node = nbase + l15;
    int zi = z[node];
    const float* pp = pos + (size_t)node * 3;
    const float* er = emb + (size_t)zi * 61;

    v16h A[2];
#pragma unroll
    for (int c = 0; c < 2; ++c) {
        v16h a;
#pragma unroll
        for (int i = 0; i < 16; ++i) {
            int k = c * 32 + off + ((i < 8) ? i : (i + 8));
            float xv = (k < 3) ? pp[k] : er[k - 3];
            a[i] = (_Float16)xv;
        }
        A[c] = a;
    }

#pragma unroll
    for (int nt = 0; nt < 4; ++nt) {
        int n = nt * 16 + l15;
        v8f acc;
#pragma unroll
        for (int i = 0; i < 8; ++i) acc[i] = 0.f;
        acc = wmma_f16(A[0], load_frag_h16(sWt + n * D64 + off), acc);
        acc = wmma_f16(A[1], load_frag_h16(sWt + n * D64 + 32 + off), acc);
#pragma unroll
        for (int v = 0; v < 8; ++v) {
            int m = nbase + off + v;
            h[(size_t)m * D64 + n]   = acc[v];
            h16[(size_t)m * D64 + n] = (_Float16)acc[v];
        }
    }
}

// ---- edge message kernel: m = relu([h_s|h_d|d2]@W1+b1)@W2+b2; agg[dst]+=m

__global__ void k_edge(const _Float16* __restrict__ h16, const float* __restrict__ d2,
                       const int* __restrict__ src, const int* __restrict__ dst,
                       const float* __restrict__ W1, const float* __restrict__ b1,
                       const float* __restrict__ W2, const float* __restrict__ b2,
                       float* __restrict__ agg, int E) {
    __shared__ alignas(16) _Float16 sW1t[D64 * 128];
    __shared__ alignas(16) _Float16 sW2t[D64 * D64];
    __shared__ alignas(16) _Float16 sM1[4][16 * D64];
    __shared__ float sB1[D64], sB2[D64], sW1L[D64];

    int tid = threadIdx.x;
    for (int i = tid; i < D64 * 128; i += blockDim.x) {
        int n = i >> 7, k = i & 127;
        sW1t[i] = (_Float16)W1[k * D64 + n];
    }
    for (int i = tid; i < D64 * D64; i += blockDim.x) {
        int n = i >> 6, k = i & 63;
        sW2t[i] = (_Float16)W2[k * D64 + n];
    }
    for (int i = tid; i < D64; i += blockDim.x) {
        sB1[i] = b1[i];
        sB2[i] = b2[i];
        sW1L[i] = W1[128 * D64 + i];
    }
    __syncthreads();

    int wv = tid >> 5, lane = tid & 31;
    int l15 = lane & 15;
    int off = (lane >> 4) << 3;
    int ntiles = E >> 4;
    _Float16* myM1 = sM1[wv];

    // Loop-invariant per-lane state: biases, rank-1 weights, W2 B-fragments.
    float bn1[4], bn2[4], wl1[4];
    v16h  Wb2[4][2];
#pragma unroll
    for (int nt = 0; nt < 4; ++nt) {
        int n = nt * 16 + l15;
        bn1[nt] = sB1[n];
        bn2[nt] = sB2[n];
        wl1[nt] = sW1L[n];
        Wb2[nt][0] = load_frag_h16(sW2t + n * D64 + off);
        Wb2[nt][1] = load_frag_h16(sW2t + n * D64 + 32 + off);
    }
    const _Float16* w1row = sW1t + l15 * 128 + off;  // + nt*16*128 per n-tile

    for (int tile = blockIdx.x * 4 + wv; tile < ntiles; tile += gridDim.x * 4) {
        int ebase = tile << 4;

        int sn = src[ebase + l15];
        int dn = dst[ebase + l15];
        v16h As0 = load_frag_h16(h16 + (size_t)sn * D64 + off);
        v16h As1 = load_frag_h16(h16 + (size_t)sn * D64 + 32 + off);
        v16h Ad0 = load_frag_h16(h16 + (size_t)dn * D64 + off);
        v16h Ad1 = load_frag_h16(h16 + (size_t)dn * D64 + 32 + off);

        v4f d2a = *(const v4f*)(d2 + ebase + off);
        v4f d2b = *(const v4f*)(d2 + ebase + off + 4);
        v4i di0 = *(const v4i*)(dst + ebase + off);
        v4i di1 = *(const v4i*)(dst + ebase + off + 4);

        // GEMM1: [h_s|h_d] @ W1[0:128] + d2 * W1[128] + b1, relu -> LDS (f16)
#pragma unroll
        for (int nt = 0; nt < 4; ++nt) {
            int n = nt * 16 + l15;
            const _Float16* wr = w1row + nt * 16 * 128;
            v8f acc;
#pragma unroll
            for (int i = 0; i < 8; ++i) acc[i] = bn1[nt];
            acc = wmma_f16(As0, load_frag_h16(wr), acc);
            acc = wmma_f16(As1, load_frag_h16(wr + 32), acc);
            acc = wmma_f16(Ad0, load_frag_h16(wr + 64), acc);
            acc = wmma_f16(Ad1, load_frag_h16(wr + 96), acc);
#pragma unroll
            for (int v = 0; v < 8; ++v) {
                float dv = (v < 4) ? d2a[v] : d2b[v - 4];
                float r = fmaxf(acc[v] + dv * wl1[nt], 0.f);
                myM1[(off + v) * D64 + n] = (_Float16)r;
            }
        }

        // GEMM2: m1 @ W2 + b2 (W2 fragments register-resident), scatter to agg
        v16h a0 = load_frag_h16(myM1 + l15 * D64 + off);
        v16h a1 = load_frag_h16(myM1 + l15 * D64 + 32 + off);
#pragma unroll
        for (int nt = 0; nt < 4; ++nt) {
            int n = nt * 16 + l15;
            v8f acc;
#pragma unroll
            for (int i = 0; i < 8; ++i) acc[i] = bn2[nt];
            acc = wmma_f16(a0, Wb2[nt][0], acc);
            acc = wmma_f16(a1, Wb2[nt][1], acc);
#pragma unroll
            for (int v = 0; v < 8; ++v) {
                int g = (v < 4) ? di0[v] : di1[v - 4];
                atomAddF(agg + (size_t)g * D64 + n, acc[v]);
            }
        }
    }
}

// ---- node update: h += relu([h|agg] @ W3 + b3) --------------------------

__global__ void k_node(float* __restrict__ h, _Float16* __restrict__ h16,
                       const float* __restrict__ agg, const float* __restrict__ W3,
                       const float* __restrict__ b3, int N) {
    __shared__ alignas(16) _Float16 sW3t[D64 * 128];
    __shared__ float sB3[D64];
    int tid = threadIdx.x;
    for (int i = tid; i < D64 * 128; i += blockDim.x) {
        int n = i >> 7, k = i & 127;
        sW3t[i] = (_Float16)W3[k * D64 + n];
    }
    for (int i = tid; i < D64; i += blockDim.x) sB3[i] = b3[i];
    __syncthreads();

    int wv = tid >> 5, lane = tid & 31;
    int tile = blockIdx.x * 4 + wv;
    if (tile * 16 >= N) return;
    int nbase = tile * 16;
    int l15 = lane & 15;
    int off = (lane >> 4) << 3;

    int node = nbase + l15;
    v16h Ah0 = load_frag_h16(h16 + (size_t)node * D64 + off);
    v16h Ah1 = load_frag_h16(h16 + (size_t)node * D64 + 32 + off);
    v16h Ag0 = load_frag_f32(agg + (size_t)node * D64 + off);
    v16h Ag1 = load_frag_f32(agg + (size_t)node * D64 + 32 + off);

#pragma unroll
    for (int nt = 0; nt < 4; ++nt) {
        int n = nt * 16 + l15;
        float bn = sB3[n];
        v8f acc;
#pragma unroll
        for (int i = 0; i < 8; ++i) acc[i] = bn;
        acc = wmma_f16(Ah0, load_frag_h16(sW3t + n * 128 + off), acc);
        acc = wmma_f16(Ah1, load_frag_h16(sW3t + n * 128 + 32 + off), acc);
        acc = wmma_f16(Ag0, load_frag_h16(sW3t + n * 128 + 64 + off), acc);
        acc = wmma_f16(Ag1, load_frag_h16(sW3t + n * 128 + 96 + off), acc);
#pragma unroll
        for (int v = 0; v < 8; ++v) {
            size_t m = (size_t)(nbase + off + v) * D64 + n;
            float nv = h[m] + fmaxf(acc[v], 0.f);
            h[m] = nv;
            h16[m] = (_Float16)nv;
        }
    }
}

// ---- projection (with input relu): h_out = relu(h_in) @ Win -------------

__global__ void k_proj(const float* __restrict__ hin, const float* __restrict__ Win,
                       float* __restrict__ hout, _Float16* __restrict__ hout16, int N) {
    __shared__ alignas(16) _Float16 sWt[D64 * D64];
    int tid = threadIdx.x;
    for (int i = tid; i < D64 * D64; i += blockDim.x) {
        int n = i >> 6, k = i & 63;
        sWt[i] = (_Float16)Win[k * D64 + n];
    }
    __syncthreads();
    int wv = tid >> 5, lane = tid & 31;
    int tile = blockIdx.x * 4 + wv;
    if (tile * 16 >= N) return;
    int nbase = tile * 16;
    int l15 = lane & 15;
    int off = (lane >> 4) << 3;

    int node = nbase + l15;
    v16h A0 = load_frag_f32_relu(hin + (size_t)node * D64 + off);
    v16h A1 = load_frag_f32_relu(hin + (size_t)node * D64 + 32 + off);

#pragma unroll
    for (int nt = 0; nt < 4; ++nt) {
        int n = nt * 16 + l15;
        v8f acc;
#pragma unroll
        for (int i = 0; i < 8; ++i) acc[i] = 0.f;
        acc = wmma_f16(A0, load_frag_h16(sWt + n * D64 + off), acc);
        acc = wmma_f16(A1, load_frag_h16(sWt + n * D64 + 32 + off), acc);
#pragma unroll
        for (int v = 0; v < 8; ++v) {
            size_t m = (size_t)(nbase + off + v) * D64 + n;
            hout[m] = acc[v];
            hout16[m] = (_Float16)acc[v];
        }
    }
}

// ---- final: y = relu(h) @ Wlin + blin; pooled[batch] += y ---------------

__global__ void k_final(const float* __restrict__ h, const int* __restrict__ batch,
                        const float* __restrict__ Wlin, const float* __restrict__ blin,
                        float* __restrict__ pooled, int N) {
    __shared__ alignas(16) _Float16 sWt[32 * D64];
    __shared__ float sBl[32];
    int tid = threadIdx.x;
    for (int i = tid; i < 32 * D64; i += blockDim.x) {
        int n = i >> 6, k = i & 63;
        sWt[i] = (_Float16)Wlin[k * 32 + n];
    }
    for (int i = tid; i < 32; i += blockDim.x) sBl[i] = blin[i];
    __syncthreads();
    int wv = tid >> 5, lane = tid & 31;
    int tile = blockIdx.x * 4 + wv;
    if (tile * 16 >= N) return;
    int nbase = tile * 16;
    int l15 = lane & 15;
    int off = (lane >> 4) << 3;

    int node = nbase + l15;
    v16h A0 = load_frag_f32_relu(h + (size_t)node * D64 + off);
    v16h A1 = load_frag_f32_relu(h + (size_t)node * D64 + 32 + off);

    v4i b0 = *(const v4i*)(batch + nbase + off);
    v4i b1 = *(const v4i*)(batch + nbase + off + 4);

#pragma unroll
    for (int nt = 0; nt < 2; ++nt) {
        int n = nt * 16 + l15;
        float bn = sBl[n];
        v8f acc;
#pragma unroll
        for (int i = 0; i < 8; ++i) acc[i] = bn;
        acc = wmma_f16(A0, load_frag_h16(sWt + n * D64 + off), acc);
        acc = wmma_f16(A1, load_frag_h16(sWt + n * D64 + 32 + off), acc);
#pragma unroll
        for (int v = 0; v < 8; ++v) {
            int g = (v < 4) ? b0[v] : b1[v - 4];
            atomAddF(pooled + (size_t)g * 32 + n, acc[v]);
        }
    }
}

__global__ void k_counts(const int* __restrict__ batch, float* __restrict__ counts, int N) {
    int i = blockIdx.x * blockDim.x + threadIdx.x;
    if (i >= N) return;
    atomAddF(counts + batch[i], 1.f);
}

__global__ void k_pool_div(const float* __restrict__ pooled, const float* __restrict__ counts,
                           float* __restrict__ out, int GO) {
    int i = blockIdx.x * blockDim.x + threadIdx.x;
    if (i >= GO) return;
    out[i] = pooled[i] / fmaxf(counts[i >> 5], 1.f);
}

// ---- launcher -----------------------------------------------------------

extern "C" void kernel_launch(void* const* d_in, const int* in_sizes, int n_in,
                              void* d_out, int out_size, void* d_ws, size_t ws_size,
                              hipStream_t stream) {
    const float* pos    = (const float*)d_in[0];
    const int*   z      = (const int*)d_in[1];
    const int*   eidx   = (const int*)d_in[2];
    const int*   batch  = (const int*)d_in[3];
    const float* emb    = (const float*)d_in[4];
    const float* t1_Win = (const float*)d_in[5];
    const float* t1_W1  = (const float*)d_in[6];
    const float* t1_b1  = (const float*)d_in[7];
    const float* t1_W2  = (const float*)d_in[8];
    const float* t1_b2  = (const float*)d_in[9];
    const float* t1_W3  = (const float*)d_in[10];
    const float* t1_b3  = (const float*)d_in[11];
    const float* t2_Win = (const float*)d_in[12];
    const float* t2_W1  = (const float*)d_in[13];
    const float* t2_b1  = (const float*)d_in[14];
    const float* t2_W2  = (const float*)d_in[15];
    const float* t2_b2  = (const float*)d_in[16];
    const float* t2_W3  = (const float*)d_in[17];
    const float* t2_b3  = (const float*)d_in[18];
    const float* Wlin   = (const float*)d_in[19];
    const float* blin   = (const float*)d_in[20];

    const int N = in_sizes[1];
    const int E = in_sizes[2] / 2;
    const int G = out_size / 32;
    const int* src = eidx;
    const int* dst = eidx + E;

    // workspace carve (256B aligned regions)
    char* ws = (char*)d_ws;
    size_t o = 0;
    auto carve = [&](size_t bytes) {
        void* p = ws + o;
        o += (bytes + 255) & ~(size_t)255;
        return p;
    };
    float*    h      = (float*)carve((size_t)N * D64 * 4);
    _Float16* h16    = (_Float16*)carve((size_t)N * D64 * 2);
    float*    agg    = (float*)carve((size_t)N * D64 * 4);
    float*    d2v    = (float*)carve((size_t)E * 4);
    float*    pooled = (float*)carve((size_t)G * 32 * 4);
    float*    counts = (float*)carve((size_t)G * 4);
    (void)ws_size;

    const int node_tiles = (N + 15) / 16;
    const int node_blocks = (node_tiles + 3) / 4;
    const int edge_blocks = 1536;

    k_embed<<<node_blocks, 128, 0, stream>>>(pos, z, emb, t1_Win, h, h16, N);
    k_d2<<<(E + 255) / 256, 256, 0, stream>>>(pos, src, dst, d2v, E);

    for (int l = 0; l < LAYERS; ++l) {
        hipMemsetAsync(agg, 0, (size_t)N * D64 * 4, stream);
        k_edge<<<edge_blocks, 128, 0, stream>>>(h16, d2v, src, dst,
                                                t1_W1 + (size_t)l * 129 * D64, t1_b1 + l * D64,
                                                t1_W2 + (size_t)l * D64 * D64, t1_b2 + l * D64,
                                                agg, E);
        k_node<<<node_blocks, 128, 0, stream>>>(h, h16, agg,
                                                t1_W3 + (size_t)l * 128 * D64, t1_b3 + l * D64, N);
    }

    k_proj<<<node_blocks, 128, 0, stream>>>(h, t2_Win, h, h16, N);

    for (int l = 0; l < LAYERS; ++l) {
        hipMemsetAsync(agg, 0, (size_t)N * D64 * 4, stream);
        k_edge<<<edge_blocks, 128, 0, stream>>>(h16, d2v, src, dst,
                                                t2_W1 + (size_t)l * 129 * D64, t2_b1 + l * D64,
                                                t2_W2 + (size_t)l * D64 * D64, t2_b2 + l * D64,
                                                agg, E);
        k_node<<<node_blocks, 128, 0, stream>>>(h, h16, agg,
                                                t2_W3 + (size_t)l * 128 * D64, t2_b3 + l * D64, N);
    }

    hipMemsetAsync(pooled, 0, (size_t)G * 32 * 4, stream);
    hipMemsetAsync(counts, 0, (size_t)G * 4, stream);
    k_final<<<node_blocks, 128, 0, stream>>>(h, batch, Wlin, blin, pooled, N);
    k_counts<<<(N + 255) / 256, 256, 0, stream>>>(batch, counts, N);
    k_pool_div<<<(G * 32 + 255) / 256, 256, 0, stream>>>(pooled, counts, (float*)d_out, G * 32);
}
